// DGCNN_46351287058975
// MI455X (gfx1250) — matrix operand order repaired and checked by reading, not compiled, not run on hardware
//
#include <hip/hip_runtime.h>
#include <hip/hip_bf16.h>
#include <cstdint>

typedef __attribute__((ext_vector_type(16))) _Float16 v16h;
typedef __attribute__((ext_vector_type(8)))  _Float16 v8h;
typedef __attribute__((ext_vector_type(8)))  float    v8f;

#define EPSL   1e-5f
#define KNN    20
#define NPTS   2048
#define NB     8
#define NEGINF (-3.4e38f)
#define SSCALE (1.0f / 64.0f)   // uniform score scale (ordering-invariant), keeps -|x|^2 in f16 range

#define WMMA_F16(a, b, c) \
  __builtin_amdgcn_wmma_f32_16x16x32_f16(false, (a), false, (b), (short)0, (c), false, false)

// ---------------------------------------------------------------------------
// Async global->LDS copy (CDNA5 ISA 15.18.3, GLOBAL_LOAD_ASYNC_TO_LDS_B128,
// GV mode: LDS address taken from VDST, global address from VADDR).
// LDS offset = low 32 bits of the generic __shared__ pointer.
// Tracked with ASYNCcnt; completed via s_wait_asynccnt.
// ---------------------------------------------------------------------------
__device__ __forceinline__ void async_copy16(uint32_t ldsOff, const void* g) {
  unsigned long long ga = (unsigned long long)(uintptr_t)g;
  asm volatile("global_load_async_to_lds_b128 %0, %1, off"
               :: "v"(ldsOff), "v"(ga) : "memory");
}
__device__ __forceinline__ void async_wait0() {
  asm volatile("s_wait_asynccnt 0x0" ::: "memory");
}

// ---------------------------------------------------------------------------
// A-fragment loader: 16-bit A 16x32, contiguous k order (k = 16*half + s; B loads match):
//   lanes 0-15 : row M = lane,    halves = K[kk+0..7]   then K[kk+8..15]
//   lanes 16-31: row M = lane-16, halves = K[kk+16..23] then K[kk+24..31]
// rowBase points at K=0 of this lane's row (LDS or global; K-contiguous).
// ---------------------------------------------------------------------------
__device__ __forceinline__ v16h ldA_frag(const _Float16* rowBase, int kk, int half) {
  const _Float16* p1 = rowBase + kk + (half << 4);
  const _Float16* p2 = p1 + 8;
  v8h lo = *(const v8h*)p1;
  v8h hi = *(const v8h*)p2;
  v16h r;
#pragma unroll
  for (int i = 0; i < 8; ++i) { r[i] = lo[i]; r[i + 8] = hi[i]; }
  return r;
}

// ---------------------------------------------------------------------------
// Top-k helpers: per-lane sorted descending list + wave32 butterfly arg-max
// merge.  Tie-break matches jax.lax.top_k: value desc, index asc.
// ---------------------------------------------------------------------------
__device__ __forceinline__ void topk_insert(float s, int m, float* tv, int* ti) {
  if (s > tv[KNN - 1] || (s == tv[KNN - 1] && m < ti[KNN - 1])) {
    int pos = KNN - 1;
    while (pos > 0 && (s > tv[pos - 1] || (s == tv[pos - 1] && m < ti[pos - 1]))) {
      tv[pos] = tv[pos - 1]; ti[pos] = ti[pos - 1]; --pos;
    }
    tv[pos] = s; ti[pos] = m;
  }
}

__device__ __forceinline__ void topk_merge_write(float* tv, int* ti, int lane, int* dst) {
  int ptr = 0;
  for (int r = 0; r < KNN; ++r) {
    float bv = (ptr < KNN) ? tv[ptr] : NEGINF;
    int   bi = (ptr < KNN) ? ti[ptr] : 0x7fffffff;
    for (int off = 16; off > 0; off >>= 1) {
      float ov = __shfl_xor(bv, off, 32);
      int   oi = __shfl_xor(bi, off, 32);
      if (ov > bv || (ov == bv && oi < bi)) { bv = ov; bi = oi; }
    }
    if (ptr < KNN && tv[ptr] == bv && ti[ptr] == bi) ++ptr;  // unique winner pops
    if (lane == 0) dst[r] = bi;
  }
}

// ---------------------------------------------------------------------------
// Weight prep: fold BN scale into W, convert to f16, pad K to Kpad.
// grid = O blocks, 128 threads
// ---------------------------------------------------------------------------
__global__ __launch_bounds__(128)
void prep_weights_kernel(const float* __restrict__ W, const float* __restrict__ g,
                         const float* __restrict__ bb, const float* __restrict__ mmn,
                         const float* __restrict__ vvn, _Float16* __restrict__ Wh,
                         float* __restrict__ bias, int C2, int Kpad) {
  const int o = blockIdx.x, tid = threadIdx.x;
  const float s = g[o] * rsqrtf(vvn[o] + EPSL);
  if (tid == 0) bias[o] = bb[o] - mmn[o] * s;
  for (int c = tid; c < Kpad; c += 128) {
    float v = (c < C2) ? W[(size_t)o * C2 + c] * s : 0.f;
    Wh[(size_t)o * Kpad + c] = (_Float16)v;
  }
}

// ---------------------------------------------------------------------------
// Split-f16 prep for the WMMA distance GEMM.
//   A row (point p): [ x, 1, 0... ]                (hi/lo split)
//   B col (point p): [ 2*x*s, -|x|^2*s, 0... ]     (hi/lo split, s = SSCALE)
// score(n,m) = (2<x_n,x_m> - |x_m|^2) * s  -> same top-k ordering as fp32 pd.
// ---------------------------------------------------------------------------
__global__ __launch_bounds__(256)
void prep_split_kernel(const float* __restrict__ X,
                       _Float16* __restrict__ Ah, _Float16* __restrict__ Al,
                       _Float16* __restrict__ Bh, _Float16* __restrict__ Bl,
                       int Cin, int Kaug) {
  const int p = blockIdx.x * 256 + threadIdx.x;   // 0 .. NB*NPTS-1
  const float* x = X + (size_t)p * Cin;
  float sq = 0.f;
  for (int c = 0; c < Cin; ++c) { float v = x[c]; sq += v * v; }
  const size_t base = (size_t)p * Kaug;
  for (int c = 0; c < Kaug; ++c) {
    float av = 0.f, bv = 0.f;
    if (c < Cin)       { av = x[c];  bv = 2.f * x[c] * SSCALE; }
    else if (c == Cin) { av = 1.f;   bv = -sq * SSCALE; }
    _Float16 ah = (_Float16)av;
    _Float16 bh = (_Float16)bv;
    Ah[base + c] = ah; Al[base + c] = (_Float16)(av - (float)ah);
    Bh[base + c] = bh; Bl[base + c] = (_Float16)(bv - (float)bh);
  }
}

// ---------------------------------------------------------------------------
// KNN via WMMA distance GEMM (split-f16, f32 accumulate: acc += Ah*Bh + Ah*Bl
// + Al*Bh  ~ fp32-accurate) + fused top-20 selection.
// One workgroup (256 thr / 8 waves) per 16-point row tile; columns processed
// in two 1024-wide halves through a 64KB LDS score tile; each wave then owns
// 2 rows for per-lane insertion lists + butterfly merge.
// grid = (NPTS/16, NB)
// ---------------------------------------------------------------------------
__global__ __launch_bounds__(256)
void knn_wmma_topk_kernel(const _Float16* __restrict__ Ah, const _Float16* __restrict__ Al,
                          const _Float16* __restrict__ Bh, const _Float16* __restrict__ Bl,
                          int* __restrict__ idxOut, int Kaug) {
  __shared__ float scoreS[16 * 1024];   // 64 KB
  const int tid = threadIdx.x, lane = tid & 31, wid = tid >> 5;
  const int b = blockIdx.y, nt = blockIdx.x;
  const int row16 = lane & 15, half = lane >> 4;

  float tv0[KNN], tv1[KNN]; int ti0[KNN], ti1[KNN];
#pragma unroll
  for (int i = 0; i < KNN; ++i) {
    tv0[i] = NEGINF; ti0[i] = 0x7fffffff;
    tv1[i] = NEGINF; ti1[i] = 0x7fffffff;
  }

  const _Float16* arowH = Ah + ((size_t)b * NPTS + nt * 16 + row16) * Kaug;
  const _Float16* arowL = Al + ((size_t)b * NPTS + nt * 16 + row16) * Kaug;
  const int r0 = wid * 2, r1 = wid * 2 + 1;   // rows owned in selection phase

  for (int hb = 0; hb < 2; ++hb) {
    __syncthreads();   // protect scoreS against previous half's readers
    for (int ctl = wid; ctl < 64; ctl += 8) {
      const int colL = ctl * 16;                 // col within this half
      const int col  = hb * 1024 + colL + row16; // global point index (B side)
      const _Float16* bcolH = Bh + ((size_t)b * NPTS + col) * Kaug + (half << 4);
      const _Float16* bcolL = Bl + ((size_t)b * NPTS + col) * Kaug + (half << 4);
      v8f acc = {};
      for (int kk = 0; kk < Kaug; kk += 32) {
        v16h ah = ldA_frag(arowH, kk, half);
        v16h al = ldA_frag(arowL, kk, half);
        v16h bh = *(const v16h*)(bcolH + kk);
        v16h bl = *(const v16h*)(bcolL + kk);
        acc = WMMA_F16(ah, bh, acc);
        acc = WMMA_F16(ah, bl, acc);
        acc = WMMA_F16(al, bh, acc);
      }
      // C/D layout: VGPR j -> (M = j + 8*half, N = lane&15)
      const int rbase = half << 3;
#pragma unroll
      for (int j = 0; j < 8; ++j)
        scoreS[(rbase + j) * 1024 + colL + row16] = acc[j];
    }
    __syncthreads();
    for (int t = lane; t < 1024; t += 32) {
      const int m = hb * 1024 + t;
      topk_insert(scoreS[r0 * 1024 + t], m, tv0, ti0);
      topk_insert(scoreS[r1 * 1024 + t], m, tv1, ti1);
    }
  }

  int* dst = idxOut + ((size_t)b * NPTS + nt * 16) * KNN;
  topk_merge_write(tv0, ti0, lane, dst + (size_t)r0 * KNN);
  topk_merge_write(tv1, ti1, lane, dst + (size_t)r1 * KNN);
}

// ---------------------------------------------------------------------------
// EdgeConv via WMMA. One workgroup per point (b,n); 4 waves.
// Dual output: f32 Xout (next layer input) + f16 slice of the concatenated
// activation buffer Xcat[b][n][512] at channel offset catOff (conv4 input).
// ---------------------------------------------------------------------------
__global__ __launch_bounds__(128)
void edgeconv_wmma_kernel(const float* __restrict__ Xin, const int* __restrict__ knn,
                          const _Float16* __restrict__ Wh, const float* __restrict__ bias,
                          float* __restrict__ Xout, _Float16* __restrict__ Xcat, int catOff,
                          int Cin, int O, int Kpad, int kpLog2) {
  __shared__ _Float16 featS[32 * 512];   // up to 32 KB
  __shared__ float ctrS[256];
  __shared__ int idxS[KNN];
  const int tid = threadIdx.x;
  const int n = blockIdx.x, b = blockIdx.y;
  const size_t prow = (size_t)b * NPTS + n;

  for (int c = tid; c < Cin; c += 128) ctrS[c] = Xin[prow * Cin + c];
  if (tid < KNN) idxS[tid] = knn[prow * KNN + tid];
  __syncthreads();

  const int total = 32 * Kpad;
  const int C2 = 2 * Cin;
  for (int e = tid; e < total; e += 128) {
    int k = e >> kpLog2;
    int c = e & (Kpad - 1);
    float v = 0.f;
    if (k < KNN && c < C2) {
      if (c < Cin) v = Xin[((size_t)b * NPTS + idxS[k]) * Cin + c] - ctrS[c];
      else         v = ctrS[c - Cin];
    }
    featS[e] = (_Float16)v;
  }
  __syncthreads();

  const int wid = tid >> 5, lane = tid & 31;
  const int row = lane & 15, half = lane >> 4;
  const int numCT = O >> 4;
  for (int ct = wid; ct < numCT; ct += 4) {
    const int col = (ct << 4) + row;
    const float bias_o = bias[col];
    const _Float16* wcol = Wh + (size_t)col * Kpad + (half << 4);
    v8f acc0 = {};  // k rows 0..15
    v8f acc1 = {};  // k rows 16..31 (16..19 valid)
    for (int kk = 0; kk < Kpad; kk += 32) {
      v16h a0 = ldA_frag(featS + row * Kpad, kk, half);
      v16h a1 = ldA_frag(featS + (16 + row) * Kpad, kk, half);
      v16h bf = *(const v16h*)(wcol + kk);
      acc0 = WMMA_F16(a0, bf, acc0);
      acc1 = WMMA_F16(a1, bf, acc1);
    }
    float mx = NEGINF;
#pragma unroll
    for (int j = 0; j < 8; ++j) {
      float v = acc0[j] + bias_o; v = v > 0.f ? v : 0.2f * v; mx = fmaxf(mx, v);
    }
    if (half == 0) {  // acc1 rows k = 16+j, valid only j<4
#pragma unroll
      for (int j = 0; j < 4; ++j) {
        float v = acc1[j] + bias_o; v = v > 0.f ? v : 0.2f * v; mx = fmaxf(mx, v);
      }
    }
    mx = fmaxf(mx, __shfl_xor(mx, 16, 32));
    if (lane < 16) {
      Xout[prow * O + col] = mx;
      Xcat[prow * 512 + catOff + col] = (_Float16)mx;
    }
  }
}

// ---------------------------------------------------------------------------
// Final conv (1024 x 512 on cat activations) fused with global max/mean pool.
// grid = (16, NB): workgroup owns (batch b, 64 output cols).
// Both operands staged in LDS via async global->LDS DMA (ASYNCcnt):
//  - the 64KB weight block for this workgroup's columns, staged once
//  - each contiguous 16KB 16-point cat tile, staged per iteration
// p[b][0..1023] = max_n, p[b][1024..2047] = mean_n.
// ---------------------------------------------------------------------------
__global__ __launch_bounds__(128)
void conv4_pool_kernel(const _Float16* __restrict__ Xcat, const _Float16* __restrict__ Wh,
                       const float* __restrict__ bias, float* __restrict__ pOut) {
  __shared__ _Float16 catS[16 * 512];    // 16 KB
  __shared__ _Float16 wS[64 * 512];      // 64 KB
  const int tid = threadIdx.x;
  const int b = blockIdx.y;
  const int wid = tid >> 5, lane = tid & 31;
  const int row = lane & 15, half = lane >> 4;
  const int colBlk = blockIdx.x * 64;
  const int col = colBlk + (wid << 4) + row;   // 0..1023
  const float bias_o = bias[col];
  const uint32_t catBase = (uint32_t)(uintptr_t)catS;
  const uint32_t wBase   = (uint32_t)(uintptr_t)wS;

  // stage this workgroup's 64 weight columns (contiguous 64 KB) once
  {
    const char* src = (const char*)(Wh + (size_t)colBlk * 512);
    for (int ch = tid; ch < 4096; ch += 128)
      async_copy16(wBase + ch * 16, src + (size_t)ch * 16);
  }
  const _Float16* wcolS = wS + (size_t)((wid << 4) + row) * 512 + (half << 4);

  float rmax = NEGINF, rsum = 0.f;
  for (int nt = 0; nt < NPTS / 16; ++nt) {
    __syncthreads();                      // previous tile fully consumed
    const char* src = (const char*)(Xcat + ((size_t)b * NPTS + nt * 16) * 512);
    for (int ch = tid; ch < 1024; ch += 128)
      async_copy16(catBase + ch * 16, src + (size_t)ch * 16);
    async_wait0();                        // ASYNCcnt -> 0 (covers weight stage too)
    __syncthreads();
    v8f acc = {};
    for (int kk = 0; kk < 512; kk += 32) {
      v16h a  = ldA_frag(catS + row * 512, kk, half);
      v16h bf = *(const v16h*)(wcolS + kk);
      acc = WMMA_F16(a, bf, acc);
    }
#pragma unroll
    for (int j = 0; j < 8; ++j) {
      float v = acc[j] + bias_o; v = v > 0.f ? v : 0.2f * v;
      rmax = fmaxf(rmax, v); rsum += v;
    }
  }
  rmax = fmaxf(rmax, __shfl_xor(rmax, 16, 32));
  rsum += __shfl_xor(rsum, 16, 32);
  if (lane < 16) {
    pOut[(size_t)b * 2048 + col]        = rmax;
    pOut[(size_t)b * 2048 + 1024 + col] = rsum * (1.f / NPTS);
  }
}

// ---------------------------------------------------------------------------
// MLP head (fp32 exact). One workgroup per batch; layers chained through LDS.
// ---------------------------------------------------------------------------
__global__ __launch_bounds__(256)
void head_kernel(const float* __restrict__ p, const float* __restrict__ L1,
                 const float* __restrict__ L2, const float* __restrict__ L3,
                 const float* __restrict__ L4, float* __restrict__ out) {
  __shared__ float pS[2048];
  __shared__ float y1S[512];
  __shared__ float y2S[256];
  __shared__ float y3S[128];
  const int tid = threadIdx.x, b = blockIdx.x;
  for (int c = tid; c < 2048; c += 256) pS[c] = p[(size_t)b * 2048 + c];
  __syncthreads();
  for (int o = tid; o < 512; o += 256) {
    float a = 0.f;
    for (int c = 0; c < 2048; ++c) a += pS[c] * L1[(size_t)o * 2048 + c];
    y1S[o] = a > 0.f ? a : 0.01f * a;
  }
  __syncthreads();
  {
    float a = 0.f;
    for (int c = 0; c < 512; ++c) a += y1S[c] * L2[(size_t)tid * 512 + c];
    y2S[tid] = a > 0.f ? a : 0.01f * a;
  }
  __syncthreads();
  if (tid < 128) {
    float a = 0.f;
    for (int c = 0; c < 256; ++c) a += y2S[c] * L3[(size_t)tid * 256 + c];
    y3S[tid] = a > 0.f ? a : 0.01f * a;
  }
  __syncthreads();
  if (tid < 2) {
    float a = 0.f;
    for (int c = 0; c < 128; ++c) a += y3S[c] * L4[(size_t)tid * 128 + c];
    out[(size_t)b * 2 + tid] = a;
  }
}

// ---------------------------------------------------------------------------
extern "C" void kernel_launch(void* const* d_in, const int* in_sizes, int n_in,
                              void* d_out, int out_size, void* d_ws, size_t ws_size,
                              hipStream_t stream) {
  (void)in_sizes; (void)n_in; (void)out_size; (void)ws_size;
  const float* xyz = (const float*)d_in[0];
  const float *Wm[5], *gm[5], *bm[5], *mm[5], *vm[5];
  for (int i = 0; i < 5; ++i) {
    int base = 1 + 5 * i;
    Wm[i] = (const float*)d_in[base + 0];
    gm[i] = (const float*)d_in[base + 1];
    bm[i] = (const float*)d_in[base + 2];
    mm[i] = (const float*)d_in[base + 3];
    vm[i] = (const float*)d_in[base + 4];
  }
  const float* L1 = (const float*)d_in[26];
  const float* L2 = (const float*)d_in[27];
  const float* L3 = (const float*)d_in[28];
  const float* L4 = (const float*)d_in[29];

  // workspace carve (~90 MB total)
  char* wsp = (char*)d_ws;
  auto carve = [&](size_t bytes) {
    char* q = wsp;
    wsp += (bytes + 255) & ~(size_t)255;
    return (void*)q;
  };
  float* X0 = (float*)carve((size_t)NB * NPTS * 64 * 4);
  float* X1 = (float*)carve((size_t)NB * NPTS * 64 * 4);
  float* X2 = (float*)carve((size_t)NB * NPTS * 128 * 4);
  float* X3 = (float*)carve((size_t)NB * NPTS * 256 * 4);
  _Float16* Xcat16 = (_Float16*)carve((size_t)NB * NPTS * 512 * 2);
  int*   knnIdx = (int*)carve((size_t)NB * NPTS * KNN * 4);
  float* pB = (float*)carve((size_t)NB * 2048 * 4);

  // split-f16 distance buffers, sized for max Kaug (=288), reused per layer
  const size_t splitElems = (size_t)NB * NPTS * 288;
  _Float16* AhB = (_Float16*)carve(splitElems * 2);
  _Float16* AlB = (_Float16*)carve(splitElems * 2);
  _Float16* BhB = (_Float16*)carve(splitElems * 2);
  _Float16* BlB = (_Float16*)carve(splitElems * 2);

  static const int Oarr[5]  = {64, 64, 128, 256, 1024};
  static const int C2arr[5] = {6, 128, 256, 512, 512};
  static const int Kpad[5]  = {32, 128, 256, 512, 512};
  static const int KpLog[5] = {5, 7, 8, 9, 9};
  static const int KaugArr[4] = {32, 96, 160, 288};  // pad(Cin+1, 32)
  static const int CatOff[4]  = {0, 64, 128, 256};
  _Float16* Wh[5]; float* biasArr[5];
  for (int i = 0; i < 5; ++i) {
    Wh[i] = (_Float16*)carve((size_t)Oarr[i] * Kpad[i] * 2);
    biasArr[i] = (float*)carve((size_t)Oarr[i] * 4);
  }

  for (int i = 0; i < 5; ++i)
    prep_weights_kernel<<<dim3(Oarr[i]), dim3(128), 0, stream>>>(
        Wm[i], gm[i], bm[i], mm[i], vm[i], Wh[i], biasArr[i], C2arr[i], Kpad[i]);

  const float* Xin = xyz; int Cin = 3;
  float* Xouts[4] = {X0, X1, X2, X3};
  for (int l = 0; l < 4; ++l) {
    prep_split_kernel<<<dim3(NB * NPTS / 256), dim3(256), 0, stream>>>(
        Xin, AhB, AlB, BhB, BlB, Cin, KaugArr[l]);
    knn_wmma_topk_kernel<<<dim3(NPTS / 16, NB), dim3(256), 0, stream>>>(
        AhB, AlB, BhB, BlB, knnIdx, KaugArr[l]);
    edgeconv_wmma_kernel<<<dim3(NPTS, NB), dim3(128), 0, stream>>>(
        Xin, knnIdx, Wh[l], biasArr[l], Xouts[l], Xcat16, CatOff[l],
        Cin, Oarr[l], Kpad[l], KpLog[l]);
    Xin = Xouts[l]; Cin = Oarr[l];
  }

  conv4_pool_kernel<<<dim3(16, NB), dim3(128), 0, stream>>>(
      Xcat16, Wh[4], biasArr[4], pB);

  head_kernel<<<dim3(NB), dim3(256), 0, stream>>>(pB, L1, L2, L3, L4, (float*)d_out);
}
